// metaLinear_62740882260596
// MI455X (gfx1250) — compile-verified, matrix operand order checked by reading, not run on hardware
//
#include <hip/hip_runtime.h>

// Problem constants (from reference): x1 (4,4096,256), x2 (4,4096,64),
// W (4096,256), bvec (4096), out (4,4096,64)
#define T_TOKENS    16384   // 4 * 4096
#define K1          256     // IN1
#define J2          64      // IN2
#define NOUT        64      // OUT
#define WG_TOKENS   128     // tokens per workgroup (8 waves x 16)
#define WAVE_TOKENS 16
#define W_ELEMS     (J2 * NOUT * K1)   // 1,048,576
#define WCHUNK      (J2 * K1)          // 16,384 elems per j (64 rows x 256)

typedef __bf16 bf16_t;
typedef __attribute__((ext_vector_type(16))) __bf16 v16bf;
typedef __attribute__((ext_vector_type(8)))  __bf16 v8bf;
typedef __attribute__((ext_vector_type(4)))  __bf16 v4bf;
typedef __attribute__((ext_vector_type(8)))  float  v8f;
typedef __attribute__((ext_vector_type(4)))  float  v4f;
typedef __attribute__((ext_vector_type(4)))  int    v4i;

typedef __attribute__((address_space(1))) v4i* gptr_v4i;
typedef __attribute__((address_space(3))) v4i* lptr_v4i;

#if __has_builtin(__builtin_amdgcn_global_load_async_to_lds_b128)
#define HAS_ASYNC_LDS 1
#else
#define HAS_ASYNC_LDS 0
#endif

__device__ __forceinline__ void split_bf16(float f, bf16_t& hi, bf16_t& lo) {
    bf16_t h = (bf16_t)f;
    hi = h;
    lo = (bf16_t)(f - (float)h);
}

__device__ __forceinline__ v16bf cat8(v8bf a, v8bf b) {
    return __builtin_shufflevector(a, b, 0,1,2,3,4,5,6,7,8,9,10,11,12,13,14,15);
}

__device__ __forceinline__ void wait_async_lds() {
#if HAS_ASYNC_LDS
#if __has_builtin(__builtin_amdgcn_s_wait_asynccnt)
    __builtin_amdgcn_s_wait_asynccnt(0);
#else
    asm volatile("s_wait_asynccnt 0x0" ::: "memory");
#endif
#endif
}

// Copy one 32KB bf16 W chunk (16384 elems) global -> LDS with 256 threads.
__device__ __forceinline__ void stage_chunk(const bf16_t* __restrict__ gsrc,
                                            bf16_t* ldst, int tid) {
#if HAS_ASYNC_LDS
#pragma unroll
    for (int it = 0; it < 8; ++it) {
        const int c = (tid + it * 256) * 8;   // 16B chunks
        __builtin_amdgcn_global_load_async_to_lds_b128(
            (gptr_v4i)(gsrc + c),
            (lptr_v4i)(ldst + c),
            0, 0);
    }
#else
#pragma unroll
    for (int it = 0; it < 8; ++it) {
        const int c = (tid + it * 256) * 8;
        *(v8bf*)(ldst + c) = *(const v8bf*)(gsrc + c);
    }
#endif
}

// ---------------- shared per-wave compute helpers ----------------

// Build x1 A-fragments (hi/lo split). ISA 16-bit A layout (16x32): lane half h,
// elements e<8 -> K=8h+e, e>=8 -> K=16+8h+(e-8).
__device__ __forceinline__ void build_a_frags(const float* __restrict__ x1,
                                              int trow, int h,
                                              v16bf* ahi, v16bf* alo) {
    const float* x1p = x1 + (size_t)trow * K1;
#pragma unroll
    for (int kc = 0; kc < 8; ++kc) {
        float rr[16];
        *(v4f*)&rr[0]  = *(const v4f*)(x1p + kc * 32 + 8 * h);
        *(v4f*)&rr[4]  = *(const v4f*)(x1p + kc * 32 + 8 * h + 4);
        *(v4f*)&rr[8]  = *(const v4f*)(x1p + kc * 32 + 8 * h + 16);
        *(v4f*)&rr[12] = *(const v4f*)(x1p + kc * 32 + 8 * h + 20);
#pragma unroll
        for (int e = 0; e < 16; ++e) {
            bf16_t hh, ll;
            split_bf16(rr[e], hh, ll);
            ahi[kc][e] = hh;
            alo[kc][e] = ll;
        }
    }
}

// One j-iteration of fused compute: w-tile GEMM (3-term bf16 split, 3
// independent accumulator chains) + bias + x2-scaled accumulation into yacc.
__device__ __forceinline__ void compute_j(const bf16_t* whi, const bf16_t* wlo,
                                          const float* __restrict__ bvec,
                                          const float* sj, int j, int n, int h,
                                          const v16bf* ahi, const v16bf* alo,
                                          v8f* yacc) {
#pragma unroll
    for (int ot = 0; ot < 4; ++ot) {
        v8f acc0, acc1, acc2;
#pragma unroll
        for (int e = 0; e < 8; ++e) { acc0[e] = 0.f; acc1[e] = 0.f; acc2[e] = 0.f; }

        const int row = ot * 16 + n;   // o index = B column for this lane
#pragma unroll
        for (int kc = 0; kc < 8; ++kc) {
            const bf16_t* ph = whi + row * K1 + kc * 32 + 8 * h;
            const bf16_t* pl = wlo + row * K1 + kc * 32 + 8 * h;
            v16bf bhi = cat8(*(const v8bf*)ph, *(const v8bf*)(ph + 16));
            v16bf blo = cat8(*(const v8bf*)pl, *(const v8bf*)(pl + 16));

            acc0 = __builtin_amdgcn_wmma_f32_16x16x32_bf16(
                false, ahi[kc], false, bhi, (short)0, acc0, false, false);
            acc1 = __builtin_amdgcn_wmma_f32_16x16x32_bf16(
                false, alo[kc], false, bhi, (short)0, acc1, false, false);
            acc2 = __builtin_amdgcn_wmma_f32_16x16x32_bf16(
                false, ahi[kc], false, blo, (short)0, acc2, false, false);
        }

        const float bias = bvec[j * 64 + row];
#pragma unroll
        for (int k = 0; k < 8; ++k)
            yacc[ot][k] += sj[k] * ((acc0[k] + acc1[k] + acc2[k]) + bias);
    }
}

// ---------------- main kernel, workspace path ----------------
// W pre-split to bf16 hi/lo in global ws; double-buffered async LDS staging.
__global__ __launch_bounds__(256) void meta_linear_ws_kernel(
    const float*  __restrict__ x1,
    const float*  __restrict__ x2,
    const bf16_t* __restrict__ whi_g,   // [J2*NOUT][K1] bf16 hi
    const bf16_t* __restrict__ wlo_g,   // [J2*NOUT][K1] bf16 lo
    const float*  __restrict__ bvec,
    float*        __restrict__ out) {
    __shared__ bf16_t whibuf[2][WCHUNK];   // 2 x 32 KB
    __shared__ bf16_t wlobuf[2][WCHUNK];   // 2 x 32 KB
    __shared__ float  x2s[WG_TOKENS * J2]; // 32 KB   (total 160 KB)

    const int tid    = threadIdx.x;
    const int wave   = tid >> 5;
    const int lane   = tid & 31;
    const int n      = lane & 15;
    const int h      = lane >> 4;
    const int tok_wg = blockIdx.x * WG_TOKENS;
    const int t0     = tok_wg + wave * WAVE_TOKENS;

    // stage x2 tile (128 tokens x 64 f32)
    {
        const v4f* src = (const v4f*)(x2 + (size_t)tok_wg * J2);
        v4f*       dst = (v4f*)x2s;
#pragma unroll
        for (int it = 0; it < 8; ++it)
            dst[tid + it * 256] = src[tid + it * 256];
    }

    v16bf ahi[8], alo[8];
    build_a_frags(x1, t0 + n, h, ahi, alo);

    v8f yacc[4];
#pragma unroll
    for (int ot = 0; ot < 4; ++ot)
#pragma unroll
        for (int e = 0; e < 8; ++e) yacc[ot][e] = 0.0f;

    // prologue: stage j=0 into buffer 0
    stage_chunk(whi_g, whibuf[0], tid);
    stage_chunk(wlo_g, wlobuf[0], tid);
    wait_async_lds();
    __syncthreads();

    for (int j = 0; j < J2; ++j) {
        const int buf = j & 1;
        // kick off async staging of j+1 into the other buffer
        if (j + 1 < J2) {
            stage_chunk(whi_g + (size_t)(j + 1) * WCHUNK, whibuf[buf ^ 1], tid);
            stage_chunk(wlo_g + (size_t)(j + 1) * WCHUNK, wlobuf[buf ^ 1], tid);
        }

        float sj[8];
#pragma unroll
        for (int k = 0; k < 8; ++k)
            sj[k] = x2s[(wave * WAVE_TOKENS + k + 8 * h) * J2 + j];

        compute_j(whibuf[buf], wlobuf[buf], bvec, sj, j, n, h, ahi, alo, yacc);

        if (j + 1 < J2) wait_async_lds();
        __syncthreads();
    }

#pragma unroll
    for (int ot = 0; ot < 4; ++ot)
#pragma unroll
        for (int k = 0; k < 8; ++k)
            out[(size_t)(t0 + k + 8 * h) * NOUT + ot * 16 + n] = yacc[ot][k];
}

// ---------------- fallback kernel (no/undersized workspace) ----------------
__global__ __launch_bounds__(256) void meta_linear_fb_kernel(
    const float* __restrict__ x1,
    const float* __restrict__ x2,
    const float* __restrict__ W,
    const float* __restrict__ bvec,
    float*       __restrict__ out) {
    __shared__ bf16_t whi[WCHUNK];
    __shared__ bf16_t wlo[WCHUNK];
    __shared__ float  x2s[WG_TOKENS * J2];

    const int tid    = threadIdx.x;
    const int wave   = tid >> 5;
    const int lane   = tid & 31;
    const int n      = lane & 15;
    const int h      = lane >> 4;
    const int tok_wg = blockIdx.x * WG_TOKENS;
    const int t0     = tok_wg + wave * WAVE_TOKENS;

    {
        const v4f* src = (const v4f*)(x2 + (size_t)tok_wg * J2);
        v4f*       dst = (v4f*)x2s;
#pragma unroll
        for (int it = 0; it < 8; ++it)
            dst[tid + it * 256] = src[tid + it * 256];
    }

    v16bf ahi[8], alo[8];
    build_a_frags(x1, t0 + n, h, ahi, alo);

    v8f yacc[4];
#pragma unroll
    for (int ot = 0; ot < 4; ++ot)
#pragma unroll
        for (int e = 0; e < 8; ++e) yacc[ot][e] = 0.0f;

    for (int j = 0; j < J2; ++j) {
        const float* wp = W + (size_t)j * WCHUNK;
#pragma unroll
        for (int it = 0; it < 16; ++it) {
            const int f = tid * 4 + it * 1024;
            v4f v = *(const v4f*)(wp + f);
            v4bf hv, lv;
#pragma unroll
            for (int e = 0; e < 4; ++e) {
                bf16_t hh, ll;
                split_bf16(v[e], hh, ll);
                hv[e] = hh;
                lv[e] = ll;
            }
            *(v4bf*)(whi + f) = hv;
            *(v4bf*)(wlo + f) = lv;
        }
        if (j + 1 < J2) {
            const char* np = (const char*)(W + (size_t)(j + 1) * WCHUNK);
#pragma unroll
            for (int p = 0; p < 4; ++p)
                __builtin_prefetch(np + tid * 256 + p * 64, 0, 1);
        }
        __syncthreads();

        float sj[8];
#pragma unroll
        for (int k = 0; k < 8; ++k)
            sj[k] = x2s[(wave * WAVE_TOKENS + k + 8 * h) * J2 + j];

        compute_j(whi, wlo, bvec, sj, j, n, h, ahi, alo, yacc);
        __syncthreads();
    }

#pragma unroll
    for (int ot = 0; ot < 4; ++ot)
#pragma unroll
        for (int k = 0; k < 8; ++k)
            out[(size_t)(t0 + k + 8 * h) * NOUT + ot * 16 + n] = yacc[ot][k];
}

// ---------------- prep kernel: split W into bf16 hi/lo in workspace ----------------
__global__ __launch_bounds__(256) void prep_split_w_kernel(
    const float* __restrict__ W, bf16_t* __restrict__ whi, bf16_t* __restrict__ wlo) {
    const int i = (blockIdx.x * 256 + threadIdx.x) * 4;
    v4f v = *(const v4f*)(W + i);
    v4bf hv, lv;
#pragma unroll
    for (int e = 0; e < 4; ++e) {
        bf16_t hh, ll;
        split_bf16(v[e], hh, ll);
        hv[e] = hh;
        lv[e] = ll;
    }
    *(v4bf*)(whi + i) = hv;
    *(v4bf*)(wlo + i) = lv;
}

extern "C" void kernel_launch(void* const* d_in, const int* in_sizes, int n_in,
                              void* d_out, int out_size, void* d_ws, size_t ws_size,
                              hipStream_t stream) {
    (void)in_sizes; (void)n_in; (void)out_size;
    const float* x1 = (const float*)d_in[0];
    const float* x2 = (const float*)d_in[1];
    const float* W  = (const float*)d_in[2];
    const float* bv = (const float*)d_in[3];
    float* out = (float*)d_out;

    const size_t need = (size_t)2 * W_ELEMS * sizeof(bf16_t);   // 4 MB
    dim3 grid(T_TOKENS / WG_TOKENS);  // 128 workgroups
    dim3 block(256);                  // 8 waves, wave32

    if (d_ws != nullptr && ws_size >= need) {
        bf16_t* whi_g = (bf16_t*)d_ws;
        bf16_t* wlo_g = whi_g + W_ELEMS;
        prep_split_w_kernel<<<dim3(W_ELEMS / 1024), block, 0, stream>>>(W, whi_g, wlo_g);
        meta_linear_ws_kernel<<<grid, block, 0, stream>>>(x1, x2, whi_g, wlo_g, bv, out);
    } else {
        meta_linear_fb_kernel<<<grid, block, 0, stream>>>(x1, x2, W, bv, out);
    }
}
